// PrimitiveGrouping_2439541424690
// MI455X (gfx1250) — matrix operand order repaired and checked by reading, not compiled, not run on hardware
//
#include <hip/hip_runtime.h>
#include <hip/hip_bf16.h>
#include <math.h>

typedef __attribute__((ext_vector_type(16))) __bf16 v16bf;
typedef __attribute__((ext_vector_type(8)))  float  v8f;

constexpr int Bb = 8, Nn = 16384, Gg = 64, Cc = 256, Hh = 128;
constexpr float EPS = 1e-5f;
constexpr int TPW = 4;  // 16-point tiles per wave (A-fragments stay in VGPRs)

// ---- output layout (elements, return-order concatenation) ----
constexpr size_t SOFT_OFF  = 0;                                   // [B,G,N]
constexpr size_t WXYZ_OFF  = (size_t)Bb * Gg * Nn;                // [B,G,3]
constexpr size_t GRP_OFF   = WXYZ_OFF + (size_t)Bb * Gg * 3;      // [B,N]
constexpr size_t GF_OFF    = GRP_OFF + (size_t)Bb * Nn;           // [B,G,C]
constexpr size_t SCAT_OFF  = GF_OFF + (size_t)Bb * Gg * Cc;       // [B,N,C]
constexpr size_t WFOLD_OFF = SCAT_OFF + (size_t)Bb * Nn * Cc;     // [B,G,3]

// ---- workspace layout (bytes) ----
constexpr size_t WS_W1S  = 0;        // H*4 floats (wx,wy,wz,c1)
constexpr size_t WS_C2   = 2048;     // G floats
constexpr size_t WS_W2S  = 4096;     // 16 frags * 512 bf16 (A-matrix fragments)
constexpr size_t WS_RMAX = 20480;    // B*G floats
constexpr size_t WS_RSUM = 22528;    // B*G floats
constexpr size_t WS_GRP  = 24576;    // B*N ints

// ============ 1) fold BN into weights; pre-swizzle w2 into WMMA A-fragments ============
__global__ void prep_kernel(const float* __restrict__ w1, const float* __restrict__ b1,
                            const float* __restrict__ m1, const float* __restrict__ v1,
                            const float* __restrict__ g1, const float* __restrict__ be1,
                            const float* __restrict__ w2, const float* __restrict__ b2,
                            const float* __restrict__ m2, const float* __restrict__ v2,
                            const float* __restrict__ g2, const float* __restrict__ be2,
                            float* __restrict__ ws_w1s, float* __restrict__ ws_c2,
                            __bf16* __restrict__ ws_w2s) {
  int t = threadIdx.x;  // 128 threads
  if (t < Hh) {
    float s1 = g1[t] * rsqrtf(v1[t] + EPS);
    ws_w1s[4 * t + 0] = w1[3 * t + 0] * s1;
    ws_w1s[4 * t + 1] = w1[3 * t + 1] * s1;
    ws_w1s[4 * t + 2] = w1[3 * t + 2] * s1;
    ws_w1s[4 * t + 3] = (b1[t] - m1[t]) * s1 + be1[t];
  }
  if (t < Gg) {
    float s2 = g2[t] * rsqrtf(v2[t] + EPS);
    ws_c2[t] = (b2[t] - m2[t]) * s2 + be2[t];
  }
  // A-matrix fragment layout (16-bit, 16x32): lane = row M (lanes 16-31 mirror rows,
  // K offset +8); halves j: K = (j>=8?16:0) + (lanehi?8:0) + (j&7).
  for (int idx = t; idx < 16 * 512; idx += blockDim.x) {
    int frag = idx >> 9, rem = idx & 511, lane = rem >> 4, j = rem & 15;
    int mt = frag >> 2, kc = frag & 3;
    int g = mt * 16 + (lane & 15);
    int K = ((j >= 8) ? 16 : 0) + ((lane >= 16) ? 8 : 0) + (j & 7);
    int hh = kc * 32 + K;
    float s2 = g2[g] * rsqrtf(v2[g] + EPS);
    ws_w2s[idx] = (__bf16)(w2[g * Hh + hh] * s2);
  }
}

// ============ 2) logits z[B,G,N] via WMMA bf16: 64x64 per wave, A resident in VGPRs ============
__global__ void __launch_bounds__(256) logits_kernel(
    const float* __restrict__ sphere, const float* __restrict__ w1s4f,
    const float* __restrict__ c2, const __bf16* __restrict__ w2s,
    float* __restrict__ zout) {
  int lane = threadIdx.x & 31;
  int wave = threadIdx.x >> 5;
  int wid = blockIdx.x * 8 + wave;        // 2048 wave-jobs, TPW tiles each
  int tile0 = wid * TPW;                  // tiles are 16 points; 1024 tiles/batch
  int b = tile0 >> 10;                    // TPW divides 1024 -> whole job in one batch
  int nbase = (tile0 & 1023) << 4;
  int col = lane & 15;                    // point within tile (B/C/D column)
  int kbase = (lane >= 16) ? 16 : 0;      // unused; B operand lanes 16-31 carry K+8 (via ghalf)
  int ghalf = (lane >= 16) ? 8 : 0;       // C/D layout: lanes 16-31 hold rows M+8

  // Preload all 16 A fragments (w2 pre-scaled, pre-swizzled) into registers:
  // reused across TPW tiles -> 64 WMMAs per 16 fragment loads.
  v16bf afrag[16];
#pragma unroll
  for (int f = 0; f < 16; f++)
    afrag[f] = *(const v16bf*)(w2s + (f << 9) + (lane << 4));

  const float4* w1s = (const float4*)w1s4f;

  for (int t = 0; t < TPW; t++) {
    int n0 = nbase + t * 16;
    const float* sp = sphere + (size_t)(b * Nn + n0 + col) * 3;
    float px = sp[0], py = sp[1], pz = sp[2];

    v8f acc[4];
#pragma unroll
    for (int mt = 0; mt < 4; mt++)
      acc[mt] = (v8f){0.f, 0.f, 0.f, 0.f, 0.f, 0.f, 0.f, 0.f};

#pragma unroll
    for (int kc = 0; kc < 4; kc++) {
      // build B fragment: h = relu(w1s . xyz + c1) for this lane's 16 H-channels
      v16bf bfr;
#pragma unroll
      for (int j = 0; j < 16; j++) {
        float4 w = w1s[kc * 32 + ((j & 8) << 1) + ghalf + (j & 7)];
        float hv = fmaf(w.x, px, fmaf(w.y, py, fmaf(w.z, pz, w.w)));
        hv = fmaxf(hv, 0.f);
        bfr[j] = (__bf16)hv;
      }
#pragma unroll
      for (int mt = 0; mt < 4; mt++) {
        acc[mt] = __builtin_amdgcn_wmma_f32_16x16x32_bf16(
            false, afrag[mt * 4 + kc], false, bfr, (short)0, acc[mt], false, false);
      }
    }

#pragma unroll
    for (int mt = 0; mt < 4; mt++) {
#pragma unroll
      for (int r = 0; r < 8; r++) {
        int g = mt * 16 + ghalf + r;
        zout[(size_t)(b * Gg + g) * Nn + n0 + col] = acc[mt][r] + c2[g];
      }
    }
  }
}

// ============ 3) per-(b,g) max over N ============
__global__ void rowmax_kernel(const float* __restrict__ z, float* __restrict__ rmax) {
  __shared__ float red[256];
  int bg = blockIdx.x;
  const float* row = z + (size_t)bg * Nn;
  float m = -__builtin_inff();
  for (int i = threadIdx.x; i < Nn; i += 256) m = fmaxf(m, row[i]);
  red[threadIdx.x] = m;
  __syncthreads();
  for (int s = 128; s > 0; s >>= 1) {
    if (threadIdx.x < s) red[threadIdx.x] = fmaxf(red[threadIdx.x], red[threadIdx.x + s]);
    __syncthreads();
  }
  if (threadIdx.x == 0) rmax[bg] = red[0];
}

// ============ 4) per-(b,g) exp-sum + exp-weighted sphere/shape sums ============
__global__ void rowsum_kernel(const float* __restrict__ z, const float* __restrict__ sphere,
                              const float* __restrict__ shape, const float* __restrict__ rmax,
                              float* __restrict__ rsum, float* __restrict__ out) {
  __shared__ float red[256];
  int bg = blockIdx.x;
  int b = bg >> 6;  // /G
  const float* row = z + (size_t)bg * Nn;
  float m = rmax[bg];
  float se = 0.f, a0 = 0.f, a1 = 0.f, a2 = 0.f, f0 = 0.f, f1 = 0.f, f2 = 0.f;
  for (int i = threadIdx.x; i < Nn; i += 256) {
    float e = expf(row[i] - m);
    se += e;
    const float* sp = sphere + (size_t)(b * Nn + i) * 3;
    const float* sh = shape + (size_t)(b * Nn + i) * 3;
    a0 += e * sp[0]; a1 += e * sp[1]; a2 += e * sp[2];
    f0 += e * sh[0]; f1 += e * sh[1]; f2 += e * sh[2];
  }
  float vals[7] = {se, a0, a1, a2, f0, f1, f2};
  float res[7];
  for (int k = 0; k < 7; k++) {
    red[threadIdx.x] = vals[k];
    __syncthreads();
    for (int s = 128; s > 0; s >>= 1) {
      if (threadIdx.x < s) red[threadIdx.x] += red[threadIdx.x + s];
      __syncthreads();
    }
    res[k] = red[0];
    __syncthreads();
  }
  if (threadIdx.x == 0) {
    rsum[bg] = res[0];
    float inv = 1.f / res[0];
    out[WXYZ_OFF + (size_t)bg * 3 + 0] = res[1] * inv;
    out[WXYZ_OFF + (size_t)bg * 3 + 1] = res[2] * inv;
    out[WXYZ_OFF + (size_t)bg * 3 + 2] = res[3] * inv;
    out[WFOLD_OFF + (size_t)bg * 3 + 0] = res[4] * inv;
    out[WFOLD_OFF + (size_t)bg * 3 + 1] = res[5] * inv;
    out[WFOLD_OFF + (size_t)bg * 3 + 2] = res[6] * inv;
  }
}

// ============ 5) argmax over G per point ============
__global__ void argmax_kernel(const float* __restrict__ z, int* __restrict__ grp,
                              float* __restrict__ outGrp) {
  int idx = blockIdx.x * 256 + threadIdx.x;  // B*N
  int b = idx / Nn, n = idx % Nn;
  const float* base = z + (size_t)b * Gg * Nn + n;
  float best = base[0];
  int bg = 0;
  for (int g = 1; g < Gg; g++) {
    float v = base[(size_t)g * Nn];
    if (v > best) { best = v; bg = g; }
  }
  grp[idx] = bg;
  outGrp[idx] = (float)bg;
}

// ============ 6) softmax in place (z -> soft) ============
__global__ void soft_kernel(float* __restrict__ z, const float* __restrict__ rmax,
                            const float* __restrict__ rsum) {
  size_t idx = ((size_t)blockIdx.x * 256 + threadIdx.x) * 4;
  int bg = (int)(idx / Nn);  // N % 4 == 0 -> all 4 elems share bg
  float m = rmax[bg], inv = 1.f / rsum[bg];
  float4 v = *(float4*)(z + idx);
  v.x = expf(v.x - m) * inv;
  v.y = expf(v.y - m) * inv;
  v.z = expf(v.z - m) * inv;
  v.w = expf(v.w - m) * inv;
  *(float4*)(z + idx) = v;
}

// ============ 7) group_features: zero then scatter-max ============
__global__ void zero_gf_kernel(float* __restrict__ gf) {
  gf[(size_t)blockIdx.x * 256 + threadIdx.x] = 0.f;
}

__global__ void scatter_kernel(const float* __restrict__ feat, const int* __restrict__ grp,
                               unsigned int* __restrict__ gf) {
  size_t idx = (size_t)blockIdx.x * 256 + threadIdx.x;  // B*N*C
  size_t pt = idx / Cc;
  int c = (int)(idx % Cc);
  int b = (int)(pt / Nn);
  int g = grp[pt];
  // reference scatter-maxes onto zeros -> results >= 0; uint compare is monotone there
  float v = fmaxf(feat[idx], 0.f);
  atomicMax(&gf[(size_t)(b * Gg + g) * Cc + c], __float_as_uint(v));
}

// ============ 8) gather scattered_features ============
__global__ void gather_kernel(const float* __restrict__ gf, const int* __restrict__ grp,
                              float* __restrict__ scat) {
  size_t idx = ((size_t)blockIdx.x * 256 + threadIdx.x) * 4;  // B*N*C
  size_t pt = idx / Cc;
  int c = (int)(idx % Cc);
  int b = (int)(pt / Nn);
  int g = grp[pt];
  float4 v = *(const float4*)(gf + (size_t)(b * Gg + g) * Cc + c);
  *(float4*)(scat + idx) = v;
}

extern "C" void kernel_launch(void* const* d_in, const int* in_sizes, int n_in,
                              void* d_out, int out_size, void* d_ws, size_t ws_size,
                              hipStream_t stream) {
  const float* sphere   = (const float*)d_in[0];
  const float* shape    = (const float*)d_in[1];
  const float* features = (const float*)d_in[2];
  const float* w1  = (const float*)d_in[3];
  const float* b1  = (const float*)d_in[4];
  const float* m1  = (const float*)d_in[5];
  const float* v1  = (const float*)d_in[6];
  const float* g1  = (const float*)d_in[7];
  const float* be1 = (const float*)d_in[8];
  const float* w2  = (const float*)d_in[9];
  const float* b2  = (const float*)d_in[10];
  const float* m2  = (const float*)d_in[11];
  const float* v2  = (const float*)d_in[12];
  const float* g2  = (const float*)d_in[13];
  const float* be2 = (const float*)d_in[14];

  float* out = (float*)d_out;
  char* ws = (char*)d_ws;
  float*  ws_w1s = (float*)(ws + WS_W1S);
  float*  ws_c2  = (float*)(ws + WS_C2);
  __bf16* ws_w2s = (__bf16*)(ws + WS_W2S);
  float*  ws_rmax = (float*)(ws + WS_RMAX);
  float*  ws_rsum = (float*)(ws + WS_RSUM);
  int*    ws_grp  = (int*)(ws + WS_GRP);

  float* z = out + SOFT_OFF;  // logits written here, softmaxed in place later

  prep_kernel<<<1, 128, 0, stream>>>(w1, b1, m1, v1, g1, be1, w2, b2, m2, v2, g2, be2,
                                     ws_w1s, ws_c2, ws_w2s);

  // 2048 wave-jobs (4 tiles each), 8 waves (256 threads) per block
  logits_kernel<<<256, 256, 0, stream>>>(sphere, ws_w1s, ws_c2, ws_w2s, z);

  rowmax_kernel<<<Bb * Gg, 256, 0, stream>>>(z, ws_rmax);
  rowsum_kernel<<<Bb * Gg, 256, 0, stream>>>(z, sphere, shape, ws_rmax, ws_rsum, out);

  argmax_kernel<<<(Bb * Nn) / 256, 256, 0, stream>>>(z, ws_grp, out + GRP_OFF);

  soft_kernel<<<(int)((size_t)Bb * Gg * Nn / 1024), 256, 0, stream>>>(z, ws_rmax, ws_rsum);

  zero_gf_kernel<<<(Bb * Gg * Cc) / 256, 256, 0, stream>>>(out + GF_OFF);
  scatter_kernel<<<(int)((size_t)Bb * Nn * Cc / 256), 256, 0, stream>>>(
      features, ws_grp, (unsigned int*)(out + GF_OFF));
  gather_kernel<<<(int)((size_t)Bb * Nn * Cc / 1024), 256, 0, stream>>>(
      out + GF_OFF, ws_grp, out + SCAT_OFF);
}
